// Attention_59012850647315
// MI455X (gfx1250) — matrix operand order repaired and checked by reading, not compiled, not run on hardware
//
#include <hip/hip_runtime.h>
#include <hip/hip_bf16.h>

#define BB 2
#define TT 2048
#define DD 1024
#define OUTD 1024
#define HH 16
#define DEPTH 64

typedef __attribute__((ext_vector_type(16))) __bf16 v16bf;
typedef __attribute__((ext_vector_type(8)))  __bf16 bf16x8;
typedef __attribute__((ext_vector_type(8)))  float  v8f;

#define WMMA_BF16(a, b, c) \
  __builtin_amdgcn_wmma_f32_16x16x32_bf16(false, (a), false, (b), (short)0, (c), false, false)

static __device__ inline v16bf frag_cat(bf16x8 lo, bf16x8 hi) {
  v16bf f;
#pragma unroll
  for (int i = 0; i < 8; ++i) { f[i] = lo[i]; f[i + 8] = hi[i]; }
  return f;
}

// A fragment: 16(M) x 32(K), source row-major p[row][k] with given element stride.
// ISA layout: lanes 0-15 -> M=lane, elems 0..7 => K 0..7, elems 8..15 => K 16..23;
// lanes 16-31 shift K by +8.
static __device__ inline v16bf load_frag_a(const __bf16* __restrict__ p, int stride,
                                           int row, int k0) {
  int lane = threadIdx.x & 31;
  const __bf16* base = p + (size_t)(row + (lane & 15)) * stride + k0 + ((lane >> 4) ? 8 : 0);
  bf16x8 lo = *reinterpret_cast<const bf16x8*>(base);
  bf16x8 hi = *reinterpret_cast<const bf16x8*>(base + 16);
  return frag_cat(lo, hi);
}

// B fragment: 32(K) x 16(N), source stored K-contiguous per column: p[col][k].
// ISA layout: lane -> N = lane%16; elems 0..7 => K 0..7, elems 8..15 => K 16..23; lanes 16-31 shift K by +8.
static __device__ inline v16bf load_frag_b(const __bf16* __restrict__ p, int stride,
                                           int col, int k0) {
  int lane = threadIdx.x & 31;
  const __bf16* base = p + (size_t)(col + (lane & 15)) * stride + k0 + ((lane >> 4) ? 8 : 0);
  bf16x8 lo = *reinterpret_cast<const bf16x8*>(base);
  bf16x8 hi = *reinterpret_cast<const bf16x8*>(base + 16);
  return frag_cat(lo, hi);
}

__global__ void cvt_bf16_kernel(const float* __restrict__ in, __bf16* __restrict__ out, int n) {
  int i = blockIdx.x * blockDim.x + threadIdx.x;
  if (i < n) out[i] = (__bf16)in[i];
}

// Wt[n][k] = bf16(W[k][n]); K x N row-major input.
__global__ void transpose_bf16_kernel(const float* __restrict__ W, __bf16* __restrict__ Wt,
                                      int K, int N) {
  int i = blockIdx.x * blockDim.x + threadIdx.x;
  if (i < K * N) {
    int k = i / N, n = i % N;
    Wt[(size_t)n * K + k] = (__bf16)W[i];
  }
}

// C[M,N] = A[M,K](bf16) x Bt[N,K](bf16) + bias.
// One wave per 32x64 tile, software-pipelined: next k-step fragments are in
// flight while the current 8 WMMAs execute (partial s_wait_loadcnt).
// mode 0: f32 row-major out.   mode 1: bf16 head-split [bh][t][d], (v+bias)*scale.
// mode 2: bf16 head-split transposed [bh][d][t].
__global__ __launch_bounds__(256) void gemm_wmma_kernel(
    const __bf16* __restrict__ A, const __bf16* __restrict__ Bt,
    const float* __restrict__ bias, void* __restrict__ outp,
    int M, int N, int K, int mode, float scale) {
  int wid = blockIdx.x * (blockDim.x >> 5) + (threadIdx.x >> 5);
  int lane = threadIdx.x & 31;
  int nt64 = N >> 6;
  int row = (wid / nt64) << 5;   // 32-row tile
  int tn = (wid % nt64) << 6;    // 64-col group

  v8f acc[2][4];
#pragma unroll
  for (int m = 0; m < 2; ++m)
#pragma unroll
    for (int t = 0; t < 4; ++t)
#pragma unroll
      for (int j = 0; j < 8; ++j) acc[m][t][j] = 0.0f;

  // prologue: fragments for k=0
  v16bf a0 = load_frag_a(A, K, row, 0);
  v16bf a1 = load_frag_a(A, K, row + 16, 0);
  v16bf b0 = load_frag_b(Bt, K, tn, 0);
  v16bf b1 = load_frag_b(Bt, K, tn + 16, 0);
  v16bf b2 = load_frag_b(Bt, K, tn + 32, 0);
  v16bf b3 = load_frag_b(Bt, K, tn + 48, 0);

  for (int k0 = 0; k0 < K - 32; k0 += 32) {
    int kn = k0 + 32;
    // issue next-step loads before consuming current fragments
    v16bf na0 = load_frag_a(A, K, row, kn);
    v16bf na1 = load_frag_a(A, K, row + 16, kn);
    v16bf nb0 = load_frag_b(Bt, K, tn, kn);
    v16bf nb1 = load_frag_b(Bt, K, tn + 16, kn);
    v16bf nb2 = load_frag_b(Bt, K, tn + 32, kn);
    v16bf nb3 = load_frag_b(Bt, K, tn + 48, kn);

    acc[0][0] = WMMA_BF16(a0, b0, acc[0][0]);
    acc[1][0] = WMMA_BF16(a1, b0, acc[1][0]);
    acc[0][1] = WMMA_BF16(a0, b1, acc[0][1]);
    acc[1][1] = WMMA_BF16(a1, b1, acc[1][1]);
    acc[0][2] = WMMA_BF16(a0, b2, acc[0][2]);
    acc[1][2] = WMMA_BF16(a1, b2, acc[1][2]);
    acc[0][3] = WMMA_BF16(a0, b3, acc[0][3]);
    acc[1][3] = WMMA_BF16(a1, b3, acc[1][3]);

    a0 = na0; a1 = na1; b0 = nb0; b1 = nb1; b2 = nb2; b3 = nb3;
  }
  // epilogue: last k-step
  acc[0][0] = WMMA_BF16(a0, b0, acc[0][0]);
  acc[1][0] = WMMA_BF16(a1, b0, acc[1][0]);
  acc[0][1] = WMMA_BF16(a0, b1, acc[0][1]);
  acc[1][1] = WMMA_BF16(a1, b1, acc[1][1]);
  acc[0][2] = WMMA_BF16(a0, b2, acc[0][2]);
  acc[1][2] = WMMA_BF16(a1, b2, acc[1][2]);
  acc[0][3] = WMMA_BF16(a0, b3, acc[0][3]);
  acc[1][3] = WMMA_BF16(a1, b3, acc[1][3]);

  int nl = lane & 15, half = lane >> 4;
#pragma unroll
  for (int m = 0; m < 2; ++m) {
#pragma unroll
    for (int t = 0; t < 4; ++t) {
      int col = tn + t * 16 + nl;
      float bv = bias[col];
#pragma unroll
      for (int j = 0; j < 8; ++j) {
        int r = row + m * 16 + j + half * 8;
        float v = acc[m][t][j] + bv;
        if (mode == 0) {
          ((float*)outp)[(size_t)r * N + col] = v;
        } else {
          v *= scale;
          int bbat = r / TT, tt = r % TT;
          int h = col >> 6, d = col & 63;
          size_t bh = (size_t)(bbat * HH + h);
          if (mode == 1)
            ((__bf16*)outp)[(bh * TT + tt) * DEPTH + d] = (__bf16)v;
          else
            ((__bf16*)outp)[(bh * DEPTH + d) * TT + tt] = (__bf16)v;
        }
      }
    }
  }
}

// Flash attention: one wave per (b,h,16-query block). Online softmax over T keys,
// 32 keys per step (2 S-tiles). P re-shaped C->A layout through padded LDS.
// V fragments are loaded before the softmax VALU block so VMEM overlaps exp work.
__global__ __launch_bounds__(256) void attn_wmma_kernel(
    const __bf16* __restrict__ Qs, const __bf16* __restrict__ Ks,
    const __bf16* __restrict__ Vt, const float* __restrict__ bias,
    __bf16* __restrict__ attn_out) {
  __shared__ __attribute__((aligned(16))) __bf16 pbuf[8][16 * 40];  // 80B row stride
  int wband = threadIdx.x >> 5;
  int wid = blockIdx.x * 8 + wband;
  int lane = threadIdx.x & 31;
  int nl = lane & 15, half = lane >> 4;

  int qb = wid & 127;   // T/16 = 128 blocks
  int bh = wid >> 7;    // 0..31
  int qrow = qb << 4;

  const __bf16* Qp = Qs + (size_t)bh * TT * DEPTH;
  const __bf16* Kp = Ks + (size_t)bh * TT * DEPTH;
  const __bf16* Vp = Vt + (size_t)bh * DEPTH * TT;
  __bf16* pb = &pbuf[wband][0];

  v16bf qf0 = load_frag_a(Qp, DEPTH, qrow, 0);
  v16bf qf1 = load_frag_a(Qp, DEPTH, qrow, 32);

  float rm[8], rl[8];
  v8f acc[4];
#pragma unroll
  for (int j = 0; j < 8; ++j) { rm[j] = -3.0e38f; rl[j] = 0.0f; }
#pragma unroll
  for (int t = 0; t < 4; ++t)
#pragma unroll
    for (int j = 0; j < 8; ++j) acc[t][j] = 0.0f;

  for (int kt = 0; kt < TT; kt += 32) {
    v8f s0, s1;
#pragma unroll
    for (int j = 0; j < 8; ++j) { s0[j] = 0.0f; s1[j] = 0.0f; }
    {
      v16bf kb00 = load_frag_b(Kp, DEPTH, kt, 0);
      v16bf kb10 = load_frag_b(Kp, DEPTH, kt + 16, 0);
      s0 = WMMA_BF16(qf0, kb00, s0);
      s1 = WMMA_BF16(qf0, kb10, s1);
      v16bf kb01 = load_frag_b(Kp, DEPTH, kt, 32);
      v16bf kb11 = load_frag_b(Kp, DEPTH, kt + 16, 32);
      s0 = WMMA_BF16(qf1, kb01, s0);
      s1 = WMMA_BF16(qf1, kb11, s1);
    }

    // V fragments for this key chunk: independent of softmax -> issue early.
    v16bf vf0 = load_frag_b(Vp, TT, 0, kt);
    v16bf vf1 = load_frag_b(Vp, TT, 16, kt);
    v16bf vf2 = load_frag_b(Vp, TT, 32, kt);
    v16bf vf3 = load_frag_b(Vp, TT, 48, kt);

    float b0 = bias[kt + nl];        // bias broadcasts over key axis
    float b1 = bias[kt + 16 + nl];

    float alpha[8];
#pragma unroll
    for (int j = 0; j < 8; ++j) {
      float v0 = s0[j] + b0;
      float v1 = s1[j] + b1;
      float mx = fmaxf(v0, v1);
#pragma unroll
      for (int off = 1; off < 16; off <<= 1)       // row reduce inside 16-lane half
        mx = fmaxf(mx, __shfl_xor(mx, off, 32));
      float mnew = fmaxf(rm[j], mx);
      float a = __expf(rm[j] - mnew);
      float p0 = __expf(v0 - mnew);
      float p1 = __expf(v1 - mnew);
      float ps = p0 + p1;
#pragma unroll
      for (int off = 1; off < 16; off <<= 1)
        ps += __shfl_xor(ps, off, 32);
      rm[j] = mnew;
      rl[j] = rl[j] * a + ps;
      alpha[j] = a;
      int mr = j + half * 8;                       // C-layout row -> LDS row
      pb[mr * 40 + nl] = (__bf16)p0;
      pb[mr * 40 + 16 + nl] = (__bf16)p1;
    }
#pragma unroll
    for (int t = 0; t < 4; ++t)
#pragma unroll
      for (int j = 0; j < 8; ++j) acc[t][j] *= alpha[j];
    __syncthreads();

    // P as A-fragment (16x32): row = nl, elems 0..7 -> key half*8+i, 8..15 -> key 16+half*8+i
    const __bf16* prow = pb + nl * 40 + half * 8;
    bf16x8 plo = *reinterpret_cast<const bf16x8*>(prow);
    bf16x8 phi = *reinterpret_cast<const bf16x8*>(prow + 16);
    v16bf pf = frag_cat(plo, phi);

    acc[0] = WMMA_BF16(pf, vf0, acc[0]);
    acc[1] = WMMA_BF16(pf, vf1, acc[1]);
    acc[2] = WMMA_BF16(pf, vf2, acc[2]);
    acc[3] = WMMA_BF16(pf, vf3, acc[3]);
    __syncthreads();
  }

  int bbat = bh >> 4, h = bh & 15;
#pragma unroll
  for (int j = 0; j < 8; ++j) rl[j] = 1.0f / rl[j];
#pragma unroll
  for (int t = 0; t < 4; ++t) {
    int d = t * 16 + nl;
#pragma unroll
    for (int j = 0; j < 8; ++j) {
      int tt = qrow + j + half * 8;
      attn_out[((size_t)bbat * TT + tt) * OUTD + h * DEPTH + d] = (__bf16)(acc[t][j] * rl[j]);
    }
  }
}

extern "C" void kernel_launch(void* const* d_in, const int* in_sizes, int n_in,
                              void* d_out, int out_size, void* d_ws, size_t ws_size,
                              hipStream_t stream) {
  const float* x    = (const float*)d_in[0];
  const float* W_q  = (const float*)d_in[1];
  const float* b_q  = (const float*)d_in[2];
  const float* W_k  = (const float*)d_in[3];
  // b_k = d_in[4] is intentionally unused: reference uses b_q for the K projection.
  const float* W_v  = (const float*)d_in[5];
  const float* b_v  = (const float*)d_in[6];
  const float* W_o  = (const float*)d_in[7];
  const float* b_o  = (const float*)d_in[8];
  const float* bias = (const float*)d_in[9];

  const int M = BB * TT;                   // 4096 rows
  char* ws = (char*)d_ws;
  size_t off = 0;
  __bf16* x_bf   = (__bf16*)(ws + off); off += (size_t)M * DD * 2;          // 8 MB
  __bf16* wt_q   = (__bf16*)(ws + off); off += (size_t)DD * OUTD * 2;       // 2 MB
  __bf16* wt_k   = (__bf16*)(ws + off); off += (size_t)DD * OUTD * 2;
  __bf16* wt_v   = (__bf16*)(ws + off); off += (size_t)DD * OUTD * 2;
  __bf16* wt_o   = (__bf16*)(ws + off); off += (size_t)OUTD * OUTD * 2;
  __bf16* Qsplit = (__bf16*)(ws + off); off += (size_t)BB * HH * TT * DEPTH * 2;  // 8 MB
  __bf16* Ksplit = (__bf16*)(ws + off); off += (size_t)BB * HH * TT * DEPTH * 2;
  __bf16* Vtr    = (__bf16*)(ws + off); off += (size_t)BB * HH * DEPTH * TT * 2;
  __bf16* attn_b = (__bf16*)(ws + off); off += (size_t)M * OUTD * 2;

  // 1. fp32 -> bf16 conversions / weight transposes (K-contiguous B operands)
  cvt_bf16_kernel<<<(M * DD) / 256, 256, 0, stream>>>(x, x_bf, M * DD);
  transpose_bf16_kernel<<<(DD * OUTD) / 256, 256, 0, stream>>>(W_q, wt_q, DD, OUTD);
  transpose_bf16_kernel<<<(DD * OUTD) / 256, 256, 0, stream>>>(W_k, wt_k, DD, OUTD);
  transpose_bf16_kernel<<<(DD * OUTD) / 256, 256, 0, stream>>>(W_v, wt_v, DD, OUTD);
  transpose_bf16_kernel<<<(OUTD * OUTD) / 256, 256, 0, stream>>>(W_o, wt_o, OUTD, OUTD);

  // 2. projections: (M/32)*(N/64) waves / 8 per block = 256 blocks
  const int gblocks = (M / 32) * (OUTD / 64) / 8;
  const float qscale = 0.125f;  // DEPTH^-0.5
  gemm_wmma_kernel<<<gblocks, 256, 0, stream>>>(x_bf, wt_q, b_q, Qsplit, M, OUTD, DD, 1, qscale);
  gemm_wmma_kernel<<<gblocks, 256, 0, stream>>>(x_bf, wt_k, b_q, Ksplit, M, OUTD, DD, 1, 1.0f); // b_q bug
  gemm_wmma_kernel<<<gblocks, 256, 0, stream>>>(x_bf, wt_v, b_v, Vtr,    M, OUTD, DD, 2, 1.0f);

  // 3. attention: B*H*(T/16) = 4096 waves -> 512 blocks of 8 waves
  attn_wmma_kernel<<<(BB * HH * (TT / 16)) / 8, 256, 0, stream>>>(Qsplit, Ksplit, Vtr, bias, attn_b);

  // 4. output projection -> f32 d_out
  gemm_wmma_kernel<<<gblocks, 256, 0, stream>>>(attn_b, wt_o, b_o, d_out, M, OUTD, OUTD, 0, 1.0f);
}
